// MoEFusion_8134668058802
// MI455X (gfx1250) — matrix operand order repaired and checked by reading, not compiled, not run on hardware
//
#include <hip/hip_runtime.h>
#include <hip/hip_bf16.h>

// ---------------------------------------------------------------------------
// Fully-fused multimodal MoE for MI455X (gfx1250, wave32, WMMA bf16 16x16x32).
//   - Weights pre-converted to bf16, N-major, so a WMMA B-fragment is two 16B
//     global loads (disjoint per wave: N is split across the 4 waves).
//   - One block = 64 rows (4 M-tiles), 128 threads. Wave w owns columns
//     [w*32, w*32+32) of every 128-wide GEMM -> no duplicated B loads.
//   - proj (3 modalities, double-buffered f32->bf16 LDS staging) -> Xs (LDS)
//     gate softmax -> 8 experts (GEMM1 -> relu -> Hs -> GEMM2 -> gated sum)
//     -> GEMM3 (pre_w) -> relu -> head. X never touches HBM.
// ---------------------------------------------------------------------------

typedef __bf16 bf16;
typedef __attribute__((ext_vector_type(16))) bf16  v16bf;
typedef __attribute__((ext_vector_type(8)))  bf16  v8bf;
typedef __attribute__((ext_vector_type(8)))  float v8f;
typedef unsigned short u16;

#define B_ROWS   32768
#define D_IN     768
#define D_P      128
#define D_X      384
#define NEXP     8
#define TILE_M   64            // rows per block = 4 M-tiles
#define KSTEPS   (D_IN / 32)   // 24

// LDS layout (bytes)
#define LDS_XS    0                  // 64*384 bf16 = 49152
#define LDS_STG0  49152              // 64*32  bf16 = 4096
#define LDS_STG1  53248              // 64*32  bf16 = 4096
#define LDS_HS    57344              // 64*128 bf16 = 16384
#define LDS_GW    73728              // 64*8   f32  = 2048
#define LDS_TOTAL 75776
// Ps (64*64 f32) aliases LDS_XS after the last Xs read (barrier-guarded)

__device__ __forceinline__ v16bf combine16(v8bf lo, v8bf hi) {
  v16bf r;
#pragma unroll
  for (int i = 0; i < 8; ++i) { r[i] = lo[i]; r[8 + i] = hi[i]; }
  return r;
}

// A-fragment (16x32 bf16 row-major tile, element stride `stride`).
// ISA 7.12.2: lane<16 -> row=lane, K {0..7},{16..23}; lane>=16 -> K {8..15},{24..31}.
__device__ __forceinline__ v16bf load_a_frag(const bf16* tile, int stride) {
  const int lane = threadIdx.x & 31;
  const int row  = lane & 15;
  const int kb   = (lane >> 4) * 8;
  const bf16* p  = tile + row * stride + kb;
  v8bf lo = *(const v8bf*)p;
  v8bf hi = *(const v8bf*)(p + 16);
  return combine16(lo, hi);
}

// B-fragment (32x16 bf16) from N-major weights wt[N][kstride]:
// lane holds column n0+(lane&15), K = k0+(lane>>4)*8 .. +7 and +16 .. +23 (two 16B loads).
__device__ __forceinline__ v16bf load_b_frag(const bf16* wt, int kstride,
                                             int n0, int k0) {
  const int lane = threadIdx.x & 31;
  const bf16* p = wt + (size_t)(n0 + (lane & 15)) * kstride + k0 + ((lane >> 4) * 8);
  v8bf lo = *(const v8bf*)p;
  v8bf hi = *(const v8bf*)(p + 16);
  return combine16(lo, hi);
}

__device__ __forceinline__ v8f wmma_bf16(v16bf a, v16bf b, v8f c) {
  return __builtin_amdgcn_wmma_f32_16x16x32_bf16(
      false, a, false, b, (short)0, c, false, false);
}

// staging: 128 threads x 4 float4 = one 64x32 f32 tile
struct StgRegs { float4 v[4]; };

__device__ __forceinline__ void stg_load(const float* __restrict__ feat,
                                         int rowbase, int k0, int tid, StgRegs& r) {
  const int c4 = (tid & 7) * 4;
  const int r0 = tid >> 3;
#pragma unroll
  for (int it = 0; it < 4; ++it)
    r.v[it] = *(const float4*)(feat + (size_t)(rowbase + r0 + it * 16) * D_IN + k0 + c4);
}

__device__ __forceinline__ void stg_store(bf16* buf, int tid, const StgRegs& r) {
  const int c4 = (tid & 7) * 4;
  const int r0 = tid >> 3;
#pragma unroll
  for (int it = 0; it < 4; ++it) {
    bf16* d = buf + (r0 + it * 16) * 32 + c4;
    d[0] = (bf16)r.v[it].x; d[1] = (bf16)r.v[it].y;
    d[2] = (bf16)r.v[it].z; d[3] = (bf16)r.v[it].w;
  }
}

// ---------------------------------------------------------------------------
// Kernel 1: f32 [G][K][N] -> bf16 [G][N][K] (transpose + downconvert)
// ---------------------------------------------------------------------------
__global__ void MoEFusion_cvt_transpose(const float* __restrict__ src,
                                        u16* __restrict__ dstu,
                                        int G, int K, int N) {
  bf16* dst = (bf16*)dstu;
  int idx = blockIdx.x * blockDim.x + threadIdx.x;
  int total = G * K * N;
  if (idx >= total) return;
  int g = idx / (K * N);
  int rem = idx - g * (K * N);
  int k = rem / N;
  int n = rem - k * N;
  dst[(size_t)g * N * K + (size_t)n * K + k] = (bf16)src[idx];
}

// ---------------------------------------------------------------------------
// Kernel 2: fully fused proj + gate + experts + head
// ---------------------------------------------------------------------------
__global__ __launch_bounds__(128)
void MoEFusion_fused(const float* __restrict__ f0, const float* __restrict__ f1,
                     const float* __restrict__ f2,
                     const u16* __restrict__ projwtu,   // [3][128][768] bf16
                     const float* __restrict__ proj_b,  // [3][128]
                     const u16* __restrict__ w1tu,      // [8][128][384] bf16
                     const float* __restrict__ b1,      // [8][128]
                     const u16* __restrict__ w2tu,      // [8][128][128] bf16
                     const float* __restrict__ b2,      // [8][128]
                     const float* __restrict__ gate_w,  // [384][8]
                     const float* __restrict__ gate_b,  // [8]
                     const u16* __restrict__ prewtu,    // [64][128] bf16
                     const float* __restrict__ pre_b,   // [64]
                     const float* __restrict__ head_w,  // [64][2]
                     const float* __restrict__ head_b,  // [2]
                     float* __restrict__ out) {         // [B][2]
  extern __shared__ char smem[];
  bf16*  Xs   = (bf16*)(smem + LDS_XS);    // [64][384]
  bf16*  stg0 = (bf16*)(smem + LDS_STG0);  // [64][32]
  bf16*  stg1 = (bf16*)(smem + LDS_STG1);  // [64][32]
  bf16*  Hs   = (bf16*)(smem + LDS_HS);    // [64][128]
  float* Gw   = (float*)(smem + LDS_GW);   // [64][8]
  float* Ps   = (float*)smem;              // [64][64], aliases Xs late

  const bf16* projwt = (const bf16*)projwtu;
  const bf16* w1t    = (const bf16*)w1tu;
  const bf16* w2t    = (const bf16*)w2tu;
  const bf16* prewt  = (const bf16*)prewtu;

  const int tid  = threadIdx.x;
  const int wave = tid >> 5, lane = tid & 31;
  const int apart = lane >> 4;
  const int nlane = lane & 15;
  const int rowbase = blockIdx.x * TILE_M;
  const int ncol0 = wave * 32;             // wave-owned column base (128-wide)

  // ======================= phase 1: projection -> Xs =======================
  for (int m = 0; m < 3; ++m) {
    const float* feat = (m == 0) ? f0 : (m == 1) ? f1 : f2;
    const bf16* wt = projwt + (size_t)m * D_P * D_IN;

    v8f acc[4][2] = {};   // [Mtile][Ntile-of-wave]

    // prologue: stage k-step 0 into stg0
    {
      StgRegs r0;
      stg_load(feat, rowbase, 0, tid, r0);
      stg_store(stg0, tid, r0);
    }
    __syncthreads();

    for (int ks = 0; ks < KSTEPS; ++ks) {
      const int k0 = ks * 32;
      bf16* cur = (ks & 1) ? stg1 : stg0;
      bf16* nxt = (ks & 1) ? stg0 : stg1;

      // prefetch next k-step's f32 tile into regs (hidden behind WMMAs)
      StgRegs nr;
      if (ks + 1 < KSTEPS) stg_load(feat, rowbase, k0 + 32, tid, nr);

      v16bf b0 = load_b_frag(wt, D_IN, ncol0,      k0);
      v16bf b1 = load_b_frag(wt, D_IN, ncol0 + 16, k0);
#pragma unroll
      for (int mt = 0; mt < 4; ++mt) {
        v16bf a = load_a_frag(cur + mt * 16 * 32, 32);
        acc[mt][0] = wmma_bf16(a, b0, acc[mt][0]);
        acc[mt][1] = wmma_bf16(a, b1, acc[mt][1]);
      }

      if (ks + 1 < KSTEPS) stg_store(nxt, tid, nr);
      __syncthreads();   // cur readers done; nxt stores visible
    }

    // epilogue: + bias, bf16 into Xs[:, m*128 + col] (wave-owned columns)
#pragma unroll
    for (int ntl = 0; ntl < 2; ++ntl) {
      const int col = ncol0 + ntl * 16 + nlane;
      const float bias = proj_b[m * D_P + col];
#pragma unroll
      for (int mt = 0; mt < 4; ++mt) {
#pragma unroll
        for (int v = 0; v < 8; ++v) {
          const int row = mt * 16 + apart * 8 + v;
          Xs[row * D_X + m * D_P + col] = (bf16)(acc[mt][ntl][v] + bias);
        }
      }
    }
  }
  __syncthreads();   // Xs fully built

  // ======================= phase 2: gate softmax ===========================
  if (tid < 64) {
    float lg[NEXP];
#pragma unroll
    for (int e = 0; e < NEXP; ++e) lg[e] = gate_b[e];
    for (int k = 0; k < D_X; ++k) {
      const float xv = (float)Xs[tid * D_X + k];
#pragma unroll
      for (int e = 0; e < NEXP; ++e) lg[e] += xv * gate_w[k * NEXP + e];
    }
    float mx = lg[0];
#pragma unroll
    for (int e = 1; e < NEXP; ++e) mx = fmaxf(mx, lg[e]);
    float s = 0.f;
#pragma unroll
    for (int e = 0; e < NEXP; ++e) { lg[e] = __expf(lg[e] - mx); s += lg[e]; }
    const float inv = 1.0f / s;
#pragma unroll
    for (int e = 0; e < NEXP; ++e) Gw[tid * NEXP + e] = lg[e] * inv;
  }
  __syncthreads();

  // ======================= phase 3: experts ================================
  v8f facc[4][2] = {};   // gate-weighted expert sum (wave's 64x32 sub-tile)

  for (int e = 0; e < NEXP; ++e) {
    if (e + 1 < NEXP) {   // warm L2/WGP$ for next expert's panels
      __builtin_prefetch(w1t + (size_t)(e + 1) * D_P * D_X, 0, 3);
      __builtin_prefetch(w2t + (size_t)(e + 1) * D_P * D_P, 0, 3);
    }

    // ---- GEMM1: H = relu(X @ W1_e + b1_e), K=384 ----
    v8f acc1[4][2] = {};
    const bf16* W1 = w1t + (size_t)e * D_P * D_X;
    for (int k0 = 0; k0 < D_X; k0 += 32) {
      v16bf b0 = load_b_frag(W1, D_X, ncol0,      k0);
      v16bf b1f = load_b_frag(W1, D_X, ncol0 + 16, k0);
#pragma unroll
      for (int mt = 0; mt < 4; ++mt) {
        v16bf a = load_a_frag(Xs + mt * 16 * D_X + k0, D_X);
        acc1[mt][0] = wmma_bf16(a, b0,  acc1[mt][0]);
        acc1[mt][1] = wmma_bf16(a, b1f, acc1[mt][1]);
      }
    }
    // H write: wave-owned 32 columns, all 64 rows
    const float* b1e = b1 + e * D_P;
#pragma unroll
    for (int ntl = 0; ntl < 2; ++ntl) {
      const int cc = ncol0 + ntl * 16 + nlane;
      const float bias = b1e[cc];
#pragma unroll
      for (int mt = 0; mt < 4; ++mt) {
#pragma unroll
        for (int v = 0; v < 8; ++v) {
          const int rr = mt * 16 + apart * 8 + v;
          Hs[rr * D_P + cc] = (bf16)fmaxf(acc1[mt][ntl][v] + bias, 0.0f);
        }
      }
    }
    __syncthreads();   // H complete (columns come from all waves)

    // ---- GEMM2: EO = H @ W2_e + b2_e, K=128 ----
    v8f acc2[4][2] = {};
    const bf16* W2 = w2t + (size_t)e * D_P * D_P;
#pragma unroll
    for (int k0 = 0; k0 < D_P; k0 += 32) {
      v16bf b0 = load_b_frag(W2, D_P, ncol0,      k0);
      v16bf b1f = load_b_frag(W2, D_P, ncol0 + 16, k0);
#pragma unroll
      for (int mt = 0; mt < 4; ++mt) {
        v16bf a = load_a_frag(Hs + mt * 16 * D_P + k0, D_P);
        acc2[mt][0] = wmma_bf16(a, b0,  acc2[mt][0]);
        acc2[mt][1] = wmma_bf16(a, b1f, acc2[mt][1]);
      }
    }
    __syncthreads();   // all waves done reading Hs before next overwrite

    // gate-weighted accumulate
    const float* b2e = b2 + e * D_P;
#pragma unroll
    for (int ntl = 0; ntl < 2; ++ntl) {
      const int cc = ncol0 + ntl * 16 + nlane;
      const float bias = b2e[cc];
#pragma unroll
      for (int mt = 0; mt < 4; ++mt) {
#pragma unroll
        for (int v = 0; v < 8; ++v) {
          const int rr = mt * 16 + apart * 8 + v;
          facc[mt][ntl][v] += Gw[rr * NEXP + e] * (acc2[mt][ntl][v] + bias);
        }
      }
    }
  }

  // ======================= phase 4: pre + head =============================
  // stage fused (f32 -> bf16) into Hs (wave-owned columns, all rows)
#pragma unroll
  for (int ntl = 0; ntl < 2; ++ntl) {
    const int cc = ncol0 + ntl * 16 + nlane;
#pragma unroll
    for (int mt = 0; mt < 4; ++mt) {
#pragma unroll
      for (int v = 0; v < 8; ++v) {
        const int rr = mt * 16 + apart * 8 + v;
        Hs[rr * D_P + cc] = (bf16)facc[mt][ntl][v];
      }
    }
  }
  __syncthreads();

  // GEMM3: penult = relu(fused @ pre_w + pre_b), N=64 -> one 16-wide
  // N-tile per wave, K=128
  v8f pacc[4] = {};
#pragma unroll
  for (int k0 = 0; k0 < D_P; k0 += 32) {
    v16bf b = load_b_frag(prewt, D_P, wave * 16, k0);
#pragma unroll
    for (int mt = 0; mt < 4; ++mt) {
      v16bf a = load_a_frag(Hs + mt * 16 * D_P + k0, D_P);
      pacc[mt] = wmma_bf16(a, b, pacc[mt]);
    }
  }
  __syncthreads();   // everyone past all Xs reads -> safe to alias Ps over Xs
  {
    const int cc = wave * 16 + nlane;
    const float bias = pre_b[cc];
#pragma unroll
    for (int mt = 0; mt < 4; ++mt) {
#pragma unroll
      for (int v = 0; v < 8; ++v) {
        const int rr = mt * 16 + apart * 8 + v;
        Ps[rr * 64 + cc] = fmaxf(pacc[mt][v] + bias, 0.0f);
      }
    }
  }
  __syncthreads();

  // head: 128 threads = 64 rows x 2 outputs
  {
    const int r = tid >> 1, o = tid & 1;
    float s = head_b[o];
    const float* prow = Ps + r * 64;
#pragma unroll
    for (int k = 0; k < 64; ++k) s += prow[k] * head_w[k * 2 + o];
    out[(size_t)(rowbase + r) * 2 + o] = s;
  }
}

// ---------------------------------------------------------------------------
// Host launcher
// ---------------------------------------------------------------------------
extern "C" void kernel_launch(void* const* d_in, const int* in_sizes, int n_in,
                              void* d_out, int out_size, void* d_ws, size_t ws_size,
                              hipStream_t stream) {
  const float* f_text  = (const float*)d_in[0];
  const float* f_audio = (const float*)d_in[1];
  const float* f_video = (const float*)d_in[2];
  const float* proj_w  = (const float*)d_in[3];   // [3][768][128]
  const float* proj_b  = (const float*)d_in[4];
  const float* exp_w1  = (const float*)d_in[5];   // [8][384][128]
  const float* exp_b1  = (const float*)d_in[6];
  const float* exp_w2  = (const float*)d_in[7];   // [8][128][128]
  const float* exp_b2  = (const float*)d_in[8];
  const float* gate_w  = (const float*)d_in[9];   // [384][8]
  const float* gate_b  = (const float*)d_in[10];
  const float* pre_w   = (const float*)d_in[11];  // [128][64]
  const float* pre_b   = (const float*)d_in[12];
  const float* head_w  = (const float*)d_in[13];  // [64][2]
  const float* head_b  = (const float*)d_in[14];
  float* out = (float*)d_out;

  // workspace: bf16 transposed weights only (X stays in LDS)
  u16* projwt = (u16*)d_ws;                        // 3*128*768
  u16* w1t    = projwt + (size_t)3 * D_P * D_IN;   // 8*128*384
  u16* w2t    = w1t + (size_t)NEXP * D_P * D_X;    // 8*128*128
  u16* prewt  = w2t + (size_t)NEXP * D_P * D_P;    // 64*128

  // 1) weight conversion + transpose (tiny)
  {
    int n;
    n = 3 * D_IN * D_P;
    MoEFusion_cvt_transpose<<<(n + 255) / 256, 256, 0, stream>>>(proj_w, projwt, 3, D_IN, D_P);
    n = NEXP * D_X * D_P;
    MoEFusion_cvt_transpose<<<(n + 255) / 256, 256, 0, stream>>>(exp_w1, w1t, NEXP, D_X, D_P);
    n = NEXP * D_P * D_P;
    MoEFusion_cvt_transpose<<<(n + 255) / 256, 256, 0, stream>>>(exp_w2, w2t, NEXP, D_P, D_P);
    n = D_P * 64;
    MoEFusion_cvt_transpose<<<(n + 255) / 256, 256, 0, stream>>>(pre_w, prewt, 1, D_P, 64);
  }

  // 2) fused proj + gate + experts + head
  MoEFusion_fused<<<B_ROWS / TILE_M, 128, LDS_TOTAL, stream>>>(
      f_text, f_audio, f_video, projwt, proj_b,
      w1t, exp_b1, w2t, exp_b2, gate_w, gate_b,
      prewt, pre_b, head_w, head_b, out);
}
